// Attention_10926396801227
// MI455X (gfx1250) — matrix operand order repaired and checked by reading, not compiled, not run on hardware
//
#include <hip/hip_runtime.h>
#include <hip/hip_bf16.h>

typedef _Float16 f16;
typedef _Float16 half8 __attribute__((ext_vector_type(8)));
typedef _Float16 v16h  __attribute__((ext_vector_type(16)));
typedef float    v8f   __attribute__((ext_vector_type(8)));

#define HEADS 16
#define DHEAD 64
#define NSEQ  2048
#define BATCH 2
#define DIM   1024
#define INNER 1024
#define ROWS  (BATCH * NSEQ)   // 4096
#define NEGBIG (-3.0e38f)

// ---- fragment loaders (wave32 WMMA 16x16x32 f16, contiguous k order chosen here, not the ISA table order) ----
// B fragment: lane = column N (mod 16); K = laneHi*16 + e  -> 16 contiguous halves
__device__ __forceinline__ v16h ld_frag_b(const f16* p) {
  half8 lo = *(const half8*)p;
  half8 hi = *(const half8*)(p + 8);
  return __builtin_shufflevector(lo, hi, 0,1,2,3,4,5,6,7,8,9,10,11,12,13,14,15);
}
// A fragment: lane row M = lane%16; K = laneHi*16 + e -> 16 contiguous halves
__device__ __forceinline__ v16h ld_frag_a(const f16* p) {
  half8 lo = *(const half8*)p;
  half8 hi = *(const half8*)(p + 8);
  return __builtin_shufflevector(lo, hi, 0,1,2,3,4,5,6,7,8,9,10,11,12,13,14,15);
}
__device__ __forceinline__ v8f wmma16(v16h a, v16h b, v8f c) {
  return __builtin_amdgcn_wmma_f32_16x16x32_f16(false, a, false, b, (short)0, c, false, false);
}

// ---- CDNA5 async global->LDS copy (ASYNCcnt path, ISA 10 / 15.18.3) ----
// lds_off: addrspace(3) byte offset (low 32 bits of generic shared pointer)
__device__ __forceinline__ void async_b128(unsigned lds_off, const void* g) {
  asm volatile("global_load_async_to_lds_b128 %0, %1, off" : : "v"(lds_off), "v"(g) : "memory");
}
__device__ __forceinline__ void wait_async0() {
  asm volatile("s_wait_asynccnt 0" ::: "memory");
}
__device__ __forceinline__ unsigned lds_off(const void* p) {
  return (unsigned)(size_t)p;
}

// ---------------- LayerNorm: x[4096x1024] f32 -> xn f16 ----------------
__global__ void ln_kernel(const float* __restrict__ x, const float* __restrict__ gamma,
                          f16* __restrict__ xn) {
  int row = blockIdx.x;
  int t = threadIdx.x;
  const float* xr = x + (size_t)row * DIM;
  float v[4];
  float s = 0.f, ss = 0.f;
#pragma unroll
  for (int j = 0; j < 4; ++j) { v[j] = xr[t + j * 256]; s += v[j]; ss += v[j] * v[j]; }
#pragma unroll
  for (int off = 16; off > 0; off >>= 1) { s += __shfl_xor(s, off); ss += __shfl_xor(ss, off); }
  __shared__ float rs[8], rss[8], stat[2];
  int wid = t >> 5;
  if ((t & 31) == 0) { rs[wid] = s; rss[wid] = ss; }
  __syncthreads();
  if (t == 0) {
    float a = 0.f, b = 0.f;
    for (int i = 0; i < 8; ++i) { a += rs[i]; b += rss[i]; }
    float mean = a / DIM;
    float var = b / DIM - mean * mean;
    stat[0] = mean; stat[1] = rsqrtf(var + 1e-5f);
  }
  __syncthreads();
  float mean = stat[0], rstd = stat[1];
  f16* o = xn + (size_t)row * DIM;
#pragma unroll
  for (int j = 0; j < 4; ++j) {
    int c = t + j * 256;
    o[c] = (f16)((v[j] - mean) * rstd * gamma[c]);
  }
}

// ------- Transpose + f32->f16: src[K x N] -> dst[N x K] (N-major) -------
__global__ void transpose_conv(const float* __restrict__ src, f16* __restrict__ dst,
                               int K, int N) {
  __shared__ float tile[32][33];
  int nBase = blockIdx.x * 32, kBase = blockIdx.y * 32;
  int tx = threadIdx.x, ty = threadIdx.y;
#pragma unroll
  for (int i = 0; i < 32; i += 8)
    tile[ty + i][tx] = src[(size_t)(kBase + ty + i) * N + nBase + tx];
  __syncthreads();
#pragma unroll
  for (int i = 0; i < 32; i += 8)
    dst[(size_t)(nBase + ty + i) * K + kBase + tx] = (f16)tile[tx][ty + i];
}

// ---- QKV GEMM, LDS-staged async double-buffered B, fused RMS qk-norm ----
// A: xn [4096 x 1024] f16, Bt: [3072 x 1024] f16 (cols of [Wq|Wkv], N-major)
// Each wave: 32 rows x 64 cols. Block: 8 waves = 256 rows, one 64-col head slab.
__global__ void gemm_qkv(const f16* __restrict__ A, const f16* __restrict__ Bt,
                         const float* __restrict__ gq, const float* __restrict__ gk,
                         f16* __restrict__ qf, f16* __restrict__ kf, f16* __restrict__ vT) {
  int lane = threadIdx.x, lo = lane & 15, hi = lane >> 4;
  int wave = threadIdx.y;
  int tid = wave * 32 + lane;
  int rowBase = (blockIdx.y * 8 + wave) * 32;
  int colBase = blockIdx.x * 64;
  __shared__ __align__(16) f16 btile[2][64 * 32];   // [col][k] tile, ld = 32

  const f16* a0 = A + (size_t)(rowBase + lo) * DIM;
  const f16* a1 = A + (size_t)(rowBase + 16 + lo) * DIM;
  int cc = tid >> 2, part = tid & 3;                // copy role: 64 cols x 4 chunks
  async_b128(lds_off(&btile[0][0]) + (unsigned)(cc * 32 + part * 8) * 2,
             Bt + (size_t)(colBase + cc) * DIM + part * 8);
  v8f acc[2][4] = {};
  wait_async0();
  __syncthreads();
  int p = 0;
  for (int k0 = 0; k0 < DIM; k0 += 32) {
    int kn = k0 + 32;
    if (kn < DIM) {
      async_b128(lds_off(&btile[p ^ 1][0]) + (unsigned)(cc * 32 + part * 8) * 2,
                 Bt + (size_t)(colBase + cc) * DIM + kn + part * 8);
    }
    v16h af0 = ld_frag_a(a0 + k0 + hi * 16);
    v16h af1 = ld_frag_a(a1 + k0 + hi * 16);
    const f16* bt = btile[p];
#pragma unroll
    for (int t = 0; t < 4; ++t) {
      v16h bfr = ld_frag_b(bt + (t * 16 + lo) * 32 + hi * 16);
      acc[0][t] = wmma16(af0, bfr, acc[0][t]);
      acc[1][t] = wmma16(af1, bfr, acc[1][t]);
    }
    wait_async0();
    __syncthreads();
    p ^= 1;
  }
  int region = colBase >> 10;          // 0 = q, 1 = k, 2 = v
  int h = (colBase & 1023) >> 6;       // head within region
  if (region < 2) {
    const float* gam = (region == 0) ? gq : gk;
    f16* dst = (region == 0) ? qf : kf;
#pragma unroll
    for (int mi = 0; mi < 2; ++mi) {
      float scale[8];
#pragma unroll
      for (int r = 0; r < 8; ++r) {
        float ssq = 0.f;
#pragma unroll
        for (int t = 0; t < 4; ++t) { float c = acc[mi][t][r]; ssq += c * c; }
#pragma unroll
        for (int off = 1; off < 16; off <<= 1) ssq += __shfl_xor(ssq, off);
        scale[r] = 8.0f / fmaxf(sqrtf(ssq), 1e-12f);   // sqrt(DHEAD)=8
      }
#pragma unroll
      for (int t = 0; t < 4; ++t) {
        int d = t * 16 + lo;
        float g = gam[h * DHEAD + d];
#pragma unroll
        for (int r = 0; r < 8; ++r) {
          int row = rowBase + mi * 16 + r + hi * 8;
          int bb = row >> 11, i = row & (NSEQ - 1);
          dst[((size_t)(bb * HEADS + h) * NSEQ + i) * DHEAD + d] =
              (f16)(acc[mi][t][r] * scale[r] * g);
        }
      }
    }
  } else {
#pragma unroll
    for (int mi = 0; mi < 2; ++mi) {
#pragma unroll
      for (int t = 0; t < 4; ++t) {
        int d = t * 16 + lo;
#pragma unroll
        for (int r = 0; r < 8; ++r) {
          int row = rowBase + mi * 16 + r + hi * 8;
          int bb = row >> 11, i = row & (NSEQ - 1);
          vT[((size_t)(bb * HEADS + h) * DHEAD + d) * NSEQ + i] = (f16)acc[mi][t][r];
        }
      }
    }
  }
}

// ---- Flash attention: 8 waves/block share (b,h); K/V tiles async-staged in LDS ----
__global__ void attn_kernel(const f16* __restrict__ qf, const f16* __restrict__ kf,
                            const f16* __restrict__ vT, const int* __restrict__ mask,
                            f16* __restrict__ aout) {
  int lane = threadIdx.x, lo = lane & 15, hi = lane >> 4;
  int wave = threadIdx.y;
  int tid = wave * 32 + lane;
  int qt = blockIdx.x * 8 + wave;            // 0..127
  int h = blockIdx.y, b = blockIdx.z;
  const f16* qb = qf + ((size_t)(b * HEADS + h) * NSEQ + qt * 16) * DHEAD;
  const f16* kb = kf + (size_t)(b * HEADS + h) * NSEQ * DHEAD;
  const f16* vb = vT + (size_t)(b * HEADS + h) * DHEAD * NSEQ;
  const int* mrow = mask + (size_t)b * NSEQ;

  __shared__ __align__(16) f16 ktile[2][32 * 64];  // [j][d], ld = 64
  __shared__ __align__(16) f16 vtile[2][64 * 32];  // [d][j], ld = 32
  __shared__ __align__(16) f16 pbuf[8][16 * 32];
  f16* pw = pbuf[wave];

  int vrow = tid >> 2, part = tid & 3;       // V-tile copy role
  // prefetch jb = 0 into buffer 0 (K tile is 4KB contiguous in global)
  async_b128(lds_off(&ktile[0][0]) + (unsigned)tid * 16, (const char*)kb + tid * 16);
  async_b128(lds_off(&vtile[0][0]) + (unsigned)(vrow * 32 + part * 8) * 2,
             vb + (size_t)vrow * NSEQ + part * 8);

  v16h qa0 = ld_frag_a(qb + lo * DHEAD + hi * 16);
  v16h qa1 = ld_frag_a(qb + lo * DHEAD + 32 + hi * 16);

  v8f o[4] = {};
  float m[8], l[8];
#pragma unroll
  for (int r = 0; r < 8; ++r) { m[r] = NEGBIG; l[r] = 0.f; }

  wait_async0();
  __syncthreads();
  int p = 0;
  for (int jb = 0; jb < NSEQ; jb += 32) {
    int jn = jb + 32;
    if (jn < NSEQ) {
      async_b128(lds_off(&ktile[p ^ 1][0]) + (unsigned)tid * 16,
                 (const char*)(kb + (size_t)jn * DHEAD) + tid * 16);
      async_b128(lds_off(&vtile[p ^ 1][0]) + (unsigned)(vrow * 32 + part * 8) * 2,
                 vb + (size_t)vrow * NSEQ + jn + part * 8);
    }
    const f16* kt = ktile[p];
    const f16* vt = vtile[p];
    v8f c0 = {}, c1 = {};
    c0 = wmma16(qa0, ld_frag_b(kt + lo * 64 + hi * 16), c0);
    c0 = wmma16(qa1, ld_frag_b(kt + lo * 64 + 32 + hi * 16), c0);
    c1 = wmma16(qa0, ld_frag_b(kt + (16 + lo) * 64 + hi * 16), c1);
    c1 = wmma16(qa1, ld_frag_b(kt + (16 + lo) * 64 + 32 + hi * 16), c1);
    bool mk0 = mrow[jb + lo] != 0;
    bool mk1 = mrow[jb + 16 + lo] != 0;
#pragma unroll
    for (int r = 0; r < 8; ++r) {
      float s0 = mk0 ? c0[r] : NEGBIG;
      float s1 = mk1 ? c1[r] : NEGBIG;
      float mx = fmaxf(s0, s1);
#pragma unroll
      for (int off = 1; off < 16; off <<= 1) mx = fmaxf(mx, __shfl_xor(mx, off));
      float mn = fmaxf(m[r], mx);
      float alpha = __expf(m[r] - mn);
      float p0 = mk0 ? __expf(c0[r] - mn) : 0.f;
      float p1 = mk1 ? __expf(c1[r] - mn) : 0.f;
      float rsum = p0 + p1;
#pragma unroll
      for (int off = 1; off < 16; off <<= 1) rsum += __shfl_xor(rsum, off);
      l[r] = l[r] * alpha + rsum;
      m[r] = mn;
#pragma unroll
      for (int t = 0; t < 4; ++t) o[t][r] *= alpha;
      pw[(r + hi * 8) * 32 + lo]      = (f16)p0;   // P tile, row-major 16x32
      pw[(r + hi * 8) * 32 + 16 + lo] = (f16)p1;
    }
    // same-wave cross-lane LDS dependency: DS ops are in-order per wave
    asm volatile("s_wait_dscnt 0" ::: "memory");
    v16h pa = ld_frag_a(pw + lo * 32 + hi * 16);
#pragma unroll
    for (int t = 0; t < 4; ++t) {
      v16h bv = ld_frag_b(vt + (t * 16 + lo) * 32 + hi * 16);
      o[t] = wmma16(pa, bv, o[t]);
    }
    wait_async0();
    __syncthreads();
    p ^= 1;
  }
#pragma unroll
  for (int r = 0; r < 8; ++r) l[r] = 1.0f / l[r];
  f16* orow = aout + ((size_t)b * NSEQ + qt * 16) * INNER + h * DHEAD;
#pragma unroll
  for (int t = 0; t < 4; ++t) {
#pragma unroll
    for (int r = 0; r < 8; ++r) {
      orow[(size_t)(r + hi * 8) * INNER + t * 16 + lo] = (f16)(o[t][r] * l[r]);
    }
  }
}

// ---- Output projection: aout[4096x1024] f16 @ Wo -> out f32 (LDS-staged B) ----
__global__ void gemm_out(const f16* __restrict__ A, const f16* __restrict__ Bt,
                         float* __restrict__ out) {
  int lane = threadIdx.x, lo = lane & 15, hi = lane >> 4;
  int wave = threadIdx.y;
  int tid = wave * 32 + lane;
  int rowBase = (blockIdx.y * 8 + wave) * 32;
  int colBase = blockIdx.x * 64;
  __shared__ __align__(16) f16 btile[2][64 * 32];

  const f16* a0 = A + (size_t)(rowBase + lo) * INNER;
  const f16* a1 = A + (size_t)(rowBase + 16 + lo) * INNER;
  int cc = tid >> 2, part = tid & 3;
  async_b128(lds_off(&btile[0][0]) + (unsigned)(cc * 32 + part * 8) * 2,
             Bt + (size_t)(colBase + cc) * INNER + part * 8);
  v8f acc[2][4] = {};
  wait_async0();
  __syncthreads();
  int p = 0;
  for (int k0 = 0; k0 < INNER; k0 += 32) {
    int kn = k0 + 32;
    if (kn < INNER) {
      async_b128(lds_off(&btile[p ^ 1][0]) + (unsigned)(cc * 32 + part * 8) * 2,
                 Bt + (size_t)(colBase + cc) * INNER + kn + part * 8);
    }
    v16h af0 = ld_frag_a(a0 + k0 + hi * 16);
    v16h af1 = ld_frag_a(a1 + k0 + hi * 16);
    const f16* bt = btile[p];
#pragma unroll
    for (int t = 0; t < 4; ++t) {
      v16h bfr = ld_frag_b(bt + (t * 16 + lo) * 32 + hi * 16);
      acc[0][t] = wmma16(af0, bfr, acc[0][t]);
      acc[1][t] = wmma16(af1, bfr, acc[1][t]);
    }
    wait_async0();
    __syncthreads();
    p ^= 1;
  }
#pragma unroll
  for (int mi = 0; mi < 2; ++mi) {
#pragma unroll
    for (int t = 0; t < 4; ++t) {
#pragma unroll
      for (int r = 0; r < 8; ++r) {
        out[(size_t)(rowBase + mi * 16 + r + hi * 8) * DIM + colBase + t * 16 + lo] =
            acc[mi][t][r];
      }
    }
  }
}

extern "C" void kernel_launch(void* const* d_in, const int* in_sizes, int n_in,
                              void* d_out, int out_size, void* d_ws, size_t ws_size,
                              hipStream_t stream) {
  const float* x        = (const float*)d_in[0];
  const int*   mask     = (const int*)d_in[1];
  const float* gamma_ln = (const float*)d_in[2];
  const float* gamma_q  = (const float*)d_in[3];
  const float* gamma_k  = (const float*)d_in[4];
  const float* Wq       = (const float*)d_in[5];
  const float* Wkv      = (const float*)d_in[6];
  const float* Wo       = (const float*)d_in[7];
  float* out = (float*)d_out;

  // workspace layout (bytes), total 48 MB, 256B-aligned regions
  char* ws = (char*)d_ws;
  f16* xn    = (f16*)(ws + 0);          // 4096*1024*2 =  8 MB
  f16* qkvT  = (f16*)(ws + 8388608);    // 3072*1024*2 =  6 MB
  f16* woT   = (f16*)(ws + 14680064);   // 1024*1024*2 =  2 MB
  f16* qfh   = (f16*)(ws + 16777216);   // 8 MB [b,h,n,d]
  f16* kfh   = (f16*)(ws + 25165824);   // 8 MB [b,h,n,d]
  f16* vTh   = (f16*)(ws + 33554432);   // 8 MB [b,h,d,n]
  f16* aouth = (f16*)(ws + 41943040);   // 8 MB [b,n,h*d]

  ln_kernel<<<ROWS, 256, 0, stream>>>(x, gamma_ln, xn);
  transpose_conv<<<dim3(1024 / 32, 1024 / 32), dim3(32, 8), 0, stream>>>(Wq, qkvT, 1024, 1024);
  transpose_conv<<<dim3(2048 / 32, 1024 / 32), dim3(32, 8), 0, stream>>>(Wkv, qkvT + 1024 * 1024, 1024, 2048);
  transpose_conv<<<dim3(1024 / 32, 1024 / 32), dim3(32, 8), 0, stream>>>(Wo, woT, 1024, 1024);
  gemm_qkv<<<dim3(3072 / 64, ROWS / 256), dim3(32, 8), 0, stream>>>(xn, qkvT, gamma_q, gamma_k,
                                                                    qfh, kfh, vTh);
  attn_kernel<<<dim3(NSEQ / 128, HEADS, BATCH), dim3(32, 8), 0, stream>>>(qfh, kfh, vTh, mask, aouth);
  gemm_out<<<dim3(1024 / 64, ROWS / 256), dim3(32, 8), 0, stream>>>(aouth, woT, out);
}
